// QwenAttention_81363860455990
// MI455X (gfx1250) — matrix operand order repaired and checked by reading, not compiled, not run on hardware
//
#include <hip/hip_runtime.h>
#include <hip/hip_bf16.h>

// ---------------------------------------------------------------------------
// MI455X (gfx1250) fused attention:
//  - all matmuls via v_wmma_f32_16x16x32_bf16
//  - GEMM staging via global_load_async_to_lds_b128 (ASYNCcnt) + double buffer
// ---------------------------------------------------------------------------

typedef __attribute__((ext_vector_type(16))) __bf16 v16bf;
typedef __attribute__((ext_vector_type(8)))  float  v8f;

#define S_LEN  2048
#define HDIM   2048
#define NHEADS 16
#define NKVH   2
#define HEADD  128
#define KVCOLS (NKVH * HEADD)      // 256
#define KSTEP  64

union Frag16 { uint4 u[2]; v16bf v; };

__device__ __forceinline__ v8f wmma_bf16(v16bf a, v16bf b, v8f c) {
  return __builtin_amdgcn_wmma_f32_16x16x32_bf16(
      /*neg_a=*/false, a, /*neg_b=*/false, b,
      /*c_mod=*/(short)0, c, /*reuse_a=*/false, /*reuse_b=*/false);
}

// Async global->LDS copy of 16 bytes (per-lane addresses), tracked by ASYNCcnt.
__device__ __forceinline__ void async_copy_b128(unsigned lds_off,
                                                const void* gptr) {
  asm volatile("global_load_async_to_lds_b128 %0, %1, off"
               :: "v"(lds_off), "v"((unsigned long long)(uintptr_t)gptr)
               : "memory");
}
__device__ __forceinline__ void wait_async0() {
  asm volatile("s_wait_asynccnt 0x0" ::: "memory");
}

// ---------------------------------------------------------------------------
// 1) fp32 -> bf16 flat convert (for X / hidden_states)
// ---------------------------------------------------------------------------
__global__ void f32_to_bf16_kernel(const float* __restrict__ src,
                                   __hip_bfloat16* __restrict__ dst, size_t n) {
  size_t i = (size_t)blockIdx.x * blockDim.x + threadIdx.x;
  if (i < n) dst[i] = __float2bfloat16(src[i]);
}

// ---------------------------------------------------------------------------
// 2) W [K x N] fp32 row-major -> Wt [N x K] bf16 row-major (N-major)
//    so WMMA B-fragments become contiguous LDS loads.
// ---------------------------------------------------------------------------
__global__ void transpose_w_kernel(const float* __restrict__ W,
                                   __hip_bfloat16* __restrict__ Wt,
                                   int Krows, int Ncols) {
  size_t idx = (size_t)blockIdx.x * blockDim.x + threadIdx.x;
  size_t total = (size_t)Krows * Ncols;
  if (idx >= total) return;
  int k = (int)(idx % Krows);
  int n = (int)(idx / Krows);                 // writes contiguous along k
  Wt[(size_t)n * Krows + k] = __float2bfloat16(W[(size_t)k * Ncols + n]);
}

// ---------------------------------------------------------------------------
// 3) GEMM: C[M x N] fp32 = A[M x K] bf16 @ Bt[N x K]^T bf16 (+ bias)
//    128x128 block tile, 256 threads = 8 waves, each wave 2x4 16x16 tiles.
//    K staged 64 at a time via async-to-LDS with ping-pong buffers:
//      issue async(next) -> 16 WMMAs on (cur) -> s_wait_asynccnt 0 -> barrier.
// ---------------------------------------------------------------------------
__global__ __launch_bounds__(256)
void gemm_bf16_kernel(const __hip_bfloat16* __restrict__ A,
                      const __hip_bfloat16* __restrict__ Bt,
                      const float* __restrict__ bias,
                      float* __restrict__ C,
                      int M, int N, int Kd) {
  __shared__ __hip_bfloat16 As[2][128][KSTEP];
  __shared__ __hip_bfloat16 Bs[2][128][KSTEP];

  const int bm   = blockIdx.x * 128;
  const int bn   = blockIdx.y * 128;
  const int wave = threadIdx.x >> 5;
  const int lane = threadIdx.x & 31;
  const int half = lane >> 4;          // which 16-lane half of the wave
  const int ln   = lane & 15;
  const int wm   = (wave & 3) * 32;    // wave row offset in tile
  const int wn   = (wave >> 2) * 64;   // wave col offset in tile

  v8f acc[2][4];
#pragma unroll
  for (int mi = 0; mi < 2; ++mi)
#pragma unroll
    for (int ni = 0; ni < 4; ++ni) acc[mi][ni] = {};

  // staging: thread covers one 32-elem (64 B) segment of one row, per matrix
  const int tr = threadIdx.x >> 1;          // 0..127 staging row
  const int tc = (threadIdx.x & 1) * 32;    // 0 / 32 staging col

  const __hip_bfloat16* ga0 = A  + (size_t)(bm + tr) * Kd + tc;
  const __hip_bfloat16* gb0 = Bt + (size_t)(bn + tr) * Kd + tc;

  // prologue: stage buffer 0
  {
    unsigned la = (unsigned)(uintptr_t)&As[0][tr][tc];
    unsigned lb = (unsigned)(uintptr_t)&Bs[0][tr][tc];
#pragma unroll
    for (int i = 0; i < 4; ++i) {
      async_copy_b128(la + i * 16, ga0 + i * 8);
      async_copy_b128(lb + i * 16, gb0 + i * 8);
    }
  }
  wait_async0();
  __syncthreads();

  int buf = 0;
  for (int k0 = 0; k0 < Kd; k0 += KSTEP) {
    // issue async stage of the next K-slab into the other buffer
    if (k0 + KSTEP < Kd) {
      const int nb = buf ^ 1;
      unsigned la = (unsigned)(uintptr_t)&As[nb][tr][tc];
      unsigned lb = (unsigned)(uintptr_t)&Bs[nb][tr][tc];
      const __hip_bfloat16* ga = ga0 + k0 + KSTEP;
      const __hip_bfloat16* gb = gb0 + k0 + KSTEP;
#pragma unroll
      for (int i = 0; i < 4; ++i) {
        async_copy_b128(la + i * 16, ga + i * 8);
        async_copy_b128(lb + i * 16, gb + i * 8);
      }
      if (k0 + 2 * KSTEP < Kd) {   // warm L2 two slabs ahead
        __builtin_prefetch(ga0 + k0 + 2 * KSTEP, 0, 1);
        __builtin_prefetch(gb0 + k0 + 2 * KSTEP, 0, 1);
      }
    }

    // compute on current buffer: 2 k-chunks of 32 -> 16 WMMAs
#pragma unroll
    for (int kc = 0; kc < KSTEP; kc += 32) {
      Frag16 af[2], bf[4];
#pragma unroll
      for (int mi = 0; mi < 2; ++mi) {
        const int row = wm + mi * 16 + ln;
        // A layout: lanes<16 -> K {0..7,16..23}; lanes>=16 -> {8..15,24..31}
        af[mi].u[0] = *(const uint4*)&As[buf][row][kc + half * 8];
        af[mi].u[1] = *(const uint4*)&As[buf][row][kc + 16 + half * 8];
      }
#pragma unroll
      for (int ni = 0; ni < 4; ++ni) {
        const int col = wn + ni * 16 + ln;
        // B layout: lanes<16 -> K {0..7,16..23}; lanes>=16 -> {8..15,24..31}
        bf[ni].u[0] = *(const uint4*)&Bs[buf][col][kc + half * 8];
        bf[ni].u[1] = *(const uint4*)&Bs[buf][col][kc + 16 + half * 8];
      }
#pragma unroll
      for (int mi = 0; mi < 2; ++mi)
#pragma unroll
        for (int ni = 0; ni < 4; ++ni)
          acc[mi][ni] = wmma_bf16(af[mi].v, bf[ni].v, acc[mi][ni]);
    }

    wait_async0();       // next buffer fully in LDS (this wave's copies)
    __syncthreads();     // all waves done reading cur + staging next
    buf ^= 1;
  }

  // epilogue: C layout: VGPR r, lanes<16 -> M=r ; lanes>=16 -> M=r+8
#pragma unroll
  for (int ni = 0; ni < 4; ++ni) {
    const int col = bn + wn + ni * 16 + ln;
    const float bval = bias ? bias[col] : 0.0f;
#pragma unroll
    for (int mi = 0; mi < 2; ++mi) {
      const int rowbase = bm + wm + mi * 16 + half * 8;
#pragma unroll
      for (int r = 0; r < 8; ++r)
        C[(size_t)(rowbase + r) * N + col] = acc[mi][ni][r] + bval;
    }
  }
}

// ---------------------------------------------------------------------------
// 4) RoPE: src fp32 [S x ncols] (ncols = 2048 for Q, 256 for K).
//    Analytic cos/sin (position_ids == arange). Inputs rounded through bf16
//    to match reference precision (matmul was bf16). Writes post-RoPE bf16
//    and optionally the fp32 cache (cache_k).
// ---------------------------------------------------------------------------
__global__ void rope_kernel(const float* __restrict__ src,
                            __hip_bfloat16* __restrict__ dst_bf,
                            float* __restrict__ cache, int ncols) {
  const int npairs = ncols >> 1;
  int i = blockIdx.x * blockDim.x + threadIdx.x;
  if (i >= S_LEN * npairs) return;
  int s = i / npairs, p = i % npairs;
  int hh = p >> 6, d = p & 63;               // head, pair index within head
  int c1 = hh * HEADD + d, c2 = c1 + 64;
  const float kLogTheta = 13.815510558f;     // ln(1e6)
  float f = __expf(-(2.0f * (float)d / 128.0f) * kLogTheta);
  float ang = (float)s * f;
  float cs, sn;
  __sincosf(ang, &sn, &cs);                  // sn=sin, cs=cos
  float x1 = __bfloat162float(__float2bfloat16(src[(size_t)s * ncols + c1]));
  float x2 = __bfloat162float(__float2bfloat16(src[(size_t)s * ncols + c2]));
  float o1 = x1 * cs - x2 * sn;
  float o2 = x1 * sn + x2 * cs;
  if (cache) {
    cache[(size_t)s * ncols + c1] = o1;
    cache[(size_t)s * ncols + c2] = o2;
  }
  dst_bf[(size_t)s * ncols + c1] = __float2bfloat16(o1);
  dst_bf[(size_t)s * ncols + c2] = __float2bfloat16(o2);
}

// V pass: bf16-round (reference keeps V in bf16), write cache_v + v_bf.
__global__ void v_pass_kernel(const float* __restrict__ src,
                              float* __restrict__ cache,
                              __hip_bfloat16* __restrict__ dst_bf, size_t n) {
  size_t i = (size_t)blockIdx.x * blockDim.x + threadIdx.x;
  if (i >= n) return;
  __hip_bfloat16 b = __float2bfloat16(src[i]);
  dst_bf[i] = b;
  cache[i] = __bfloat162float(b);
}

// ---------------------------------------------------------------------------
// 5) Fused causal flash attention.
//    grid = (16 q-tiles, 16 heads), 256 threads = 8 waves, wave owns 16 rows.
//    Keys staged 32 at a time: K row-major in LDS, V transposed in LDS so the
//    PV B-fragment is a contiguous ds_load_b128. Online fp32 softmax with
//    width-16 __shfl_xor row reductions (wave32). P goes through LDS to
//    convert C-layout -> A-fragment layout.
// ---------------------------------------------------------------------------
__global__ __launch_bounds__(256)
void attn_kernel(const __hip_bfloat16* __restrict__ q_bf,
                 const __hip_bfloat16* __restrict__ k_bf,
                 const __hip_bfloat16* __restrict__ v_bf,
                 __hip_bfloat16* __restrict__ attn_bf) {
  __shared__ __hip_bfloat16 Ks[32][128];     // [key][dim]
  __shared__ __hip_bfloat16 Vt[128][32];     // [dim][key] (transposed)
  __shared__ __hip_bfloat16 Ps[8][16][32];   // per-wave P staging [row][key]

  const int qt   = blockIdx.x;               // q tile (128 rows)
  const int h    = blockIdx.y;               // head
  const int kvh  = h >> 3;                   // GQA: NH/NKV = 8
  const int wave = threadIdx.x >> 5;
  const int lane = threadIdx.x & 31;
  const int half = lane >> 4;
  const int ln   = lane & 15;
  const int qrow0 = qt * 128 + wave * 16;    // this wave's first query row

  // Preload Q A-fragments for all 4 K-chunks (16 rows x 128 dims bf16).
  Frag16 qa[4];
  const __hip_bfloat16* qrowp =
      q_bf + (size_t)(qrow0 + ln) * HDIM + h * HEADD;
#pragma unroll
  for (int c = 0; c < 4; ++c) {
    const int c0 = c * 32;
    qa[c].u[0] = *(const uint4*)(qrowp + c0 + half * 8);
    qa[c].u[1] = *(const uint4*)(qrowp + c0 + 16 + half * 8);
  }

  v8f O[8];
#pragma unroll
  for (int dt = 0; dt < 8; ++dt) O[dt] = {};
  float m[8], l[8];
#pragma unroll
  for (int r = 0; r < 8; ++r) { m[r] = -3.0e38f; l[r] = 0.0f; }

  const float scale = 0.088388347648318447f;   // 1/sqrt(128)
  const int kend = qt * 128 + 128;             // causal limit for this block

  for (int kb = 0; kb < kend; kb += 32) {
    // ---- cooperative stage: 32 keys of K (row-major) and V (transposed)
    {
      const int key = threadIdx.x >> 3;        // 0..31
      const int dc  = (threadIdx.x & 7) * 16;  // 16 dims per thread
      const uint4* ksrc =
          (const uint4*)(k_bf + (size_t)(kb + key) * KVCOLS + kvh * HEADD + dc);
      uint4 k0v = ksrc[0], k1v = ksrc[1];
      *(uint4*)&Ks[key][dc]     = k0v;
      *(uint4*)&Ks[key][dc + 8] = k1v;
      union { uint4 u[2]; __hip_bfloat16 hx[16]; } vr;
      const uint4* vsrc =
          (const uint4*)(v_bf + (size_t)(kb + key) * KVCOLS + kvh * HEADD + dc);
      vr.u[0] = vsrc[0];
      vr.u[1] = vsrc[1];
#pragma unroll
      for (int i = 0; i < 16; ++i) Vt[dc + i][key] = vr.hx[i];
    }
    __syncthreads();

    // ---- scores: two 16-key tiles, contraction over 128 dims (4 WMMAs each)
    v8f st[2];
#pragma unroll
    for (int jt = 0; jt < 2; ++jt) {
      v8f a = {};
      const int key = jt * 16 + ln;
#pragma unroll
      for (int c = 0; c < 4; ++c) {
        Frag16 kf;
        const int cb = c * 32 + half * 8;
        kf.u[0] = *(const uint4*)&Ks[key][cb];
        kf.u[1] = *(const uint4*)&Ks[key][cb + 16];
        a = wmma_bf16(qa[c].v, kf.v, a);
      }
      st[jt] = a;
    }

    // ---- causal mask + online softmax (fp32), row = half*8 + r across lanes
    float rowalpha[8];
#pragma unroll
    for (int r = 0; r < 8; ++r) {
      const int row = qrow0 + half * 8 + r;
      float s0 = st[0][r] * scale + (((kb + ln)      > row) ? -1.0e9f : 0.0f);
      float s1 = st[1][r] * scale + (((kb + 16 + ln) > row) ? -1.0e9f : 0.0f);
      float mx = fmaxf(s0, s1);
#pragma unroll
      for (int off = 8; off > 0; off >>= 1)
        mx = fmaxf(mx, __shfl_xor(mx, off, 16));
      const float mnew  = fmaxf(m[r], mx);
      const float alpha = __expf(m[r] - mnew);
      const float p0 = __expf(s0 - mnew);
      const float p1 = __expf(s1 - mnew);
      float sum = p0 + p1;
#pragma unroll
      for (int off = 8; off > 0; off >>= 1)
        sum += __shfl_xor(sum, off, 16);
      l[r] = l[r] * alpha + sum;
      m[r] = mnew;
      rowalpha[r] = alpha;
      Ps[wave][half * 8 + r][ln]      = __float2bfloat16(p0);
      Ps[wave][half * 8 + r][16 + ln] = __float2bfloat16(p1);
    }

    // rescale O accumulators by alpha (same row mapping as C layout)
#pragma unroll
    for (int dt = 0; dt < 8; ++dt)
#pragma unroll
      for (int r = 0; r < 8; ++r) O[dt][r] *= rowalpha[r];

    // ---- P A-fragment from LDS (C-layout -> A-layout; within-wave DS order)
    Frag16 pa;
    pa.u[0] = *(const uint4*)&Ps[wave][ln][half * 8];
    pa.u[1] = *(const uint4*)&Ps[wave][ln][16 + half * 8];

    // ---- O += P(16x32) @ V(32x128): 8 WMMAs, B from transposed V in LDS
#pragma unroll
    for (int dt = 0; dt < 8; ++dt) {
      Frag16 vf;
      const int d = dt * 16 + ln;
      vf.u[0] = *(const uint4*)&Vt[d][half * 8];
      vf.u[1] = *(const uint4*)&Vt[d][16 + half * 8];
      O[dt] = wmma_bf16(pa.v, vf.v, O[dt]);
    }
    __syncthreads();   // before next stage overwrites Ks/Vt
  }

  // ---- epilogue: O /= l, store bf16 into attn buffer [s][h*128+d]
  float linv[8];
#pragma unroll
  for (int r = 0; r < 8; ++r) linv[r] = 1.0f / l[r];
#pragma unroll
  for (int dt = 0; dt < 8; ++dt) {
#pragma unroll
    for (int r = 0; r < 8; ++r) {
      const int row = qrow0 + half * 8 + r;
      attn_bf[(size_t)row * HDIM + h * HEADD + dt * 16 + ln] =
          __float2bfloat16(O[dt][r] * linv[r]);
    }
  }
}

// ---------------------------------------------------------------------------
// Launch. Inputs: hidden, mask(unused, analytic), pos(unused, analytic),
//                 Wq, bq, Wk, bk, Wv, bv, Wo.
// d_out = [out fp32 S*H | cache_k fp32 S*256 | cache_v fp32 S*256].
// ---------------------------------------------------------------------------
extern "C" void kernel_launch(void* const* d_in, const int* in_sizes, int n_in,
                              void* d_out, int out_size, void* d_ws,
                              size_t ws_size, hipStream_t stream) {
  (void)in_sizes; (void)n_in; (void)out_size; (void)ws_size;
  const float* hidden = (const float*)d_in[0];
  const float* Wq = (const float*)d_in[3];
  const float* bq = (const float*)d_in[4];
  const float* Wk = (const float*)d_in[5];
  const float* bk = (const float*)d_in[6];
  const float* Wv = (const float*)d_in[7];
  const float* bv = (const float*)d_in[8];
  const float* Wo = (const float*)d_in[9];

  float* out     = (float*)d_out;
  float* cache_k = out + (size_t)S_LEN * HDIM;
  float* cache_v = cache_k + (size_t)S_LEN * KVCOLS;

  // Workspace carve-out (64 MiB total)
  char* p = (char*)d_ws;
  __hip_bfloat16* x_bf = (__hip_bfloat16*)p;  p += (size_t)S_LEN * HDIM * 2;
  __hip_bfloat16* wt_q = (__hip_bfloat16*)p;  p += (size_t)HDIM * HDIM * 2;
  __hip_bfloat16* wt_k = (__hip_bfloat16*)p;  p += (size_t)HDIM * KVCOLS * 2;
  __hip_bfloat16* wt_v = (__hip_bfloat16*)p;  p += (size_t)HDIM * KVCOLS * 2;
  __hip_bfloat16* wt_o = (__hip_bfloat16*)p;  p += (size_t)HDIM * HDIM * 2;
  float* q_f = (float*)p;                     p += (size_t)S_LEN * HDIM * 4;
  float* k_f = (float*)p;                     p += (size_t)S_LEN * KVCOLS * 4;
  float* v_f = (float*)p;                     p += (size_t)S_LEN * KVCOLS * 4;
  __hip_bfloat16* q_bf = (__hip_bfloat16*)p;  p += (size_t)S_LEN * HDIM * 2;
  __hip_bfloat16* k_bf = (__hip_bfloat16*)p;  p += (size_t)S_LEN * KVCOLS * 2;
  __hip_bfloat16* v_bf = (__hip_bfloat16*)p;  p += (size_t)S_LEN * KVCOLS * 2;
  __hip_bfloat16* a_bf = (__hip_bfloat16*)p;  p += (size_t)S_LEN * HDIM * 2;

  const int T = 256;

  // 1) converts / transposes
  {
    size_t n = (size_t)S_LEN * HDIM;
    f32_to_bf16_kernel<<<(unsigned)((n + T - 1) / T), T, 0, stream>>>(hidden, x_bf, n);
  }
  transpose_w_kernel<<<(unsigned)(((size_t)HDIM * HDIM + T - 1) / T), T, 0, stream>>>(Wq, wt_q, HDIM, HDIM);
  transpose_w_kernel<<<(unsigned)(((size_t)HDIM * KVCOLS + T - 1) / T), T, 0, stream>>>(Wk, wt_k, HDIM, KVCOLS);
  transpose_w_kernel<<<(unsigned)(((size_t)HDIM * KVCOLS + T - 1) / T), T, 0, stream>>>(Wv, wt_v, HDIM, KVCOLS);
  transpose_w_kernel<<<(unsigned)(((size_t)HDIM * HDIM + T - 1) / T), T, 0, stream>>>(Wo, wt_o, HDIM, HDIM);

  // 2) QKV projections (fp32 out + bias)
  gemm_bf16_kernel<<<dim3(S_LEN / 128, HDIM / 128),   T, 0, stream>>>(x_bf, wt_q, bq, q_f, S_LEN, HDIM,   HDIM);
  gemm_bf16_kernel<<<dim3(S_LEN / 128, KVCOLS / 128), T, 0, stream>>>(x_bf, wt_k, bk, k_f, S_LEN, KVCOLS, HDIM);
  gemm_bf16_kernel<<<dim3(S_LEN / 128, KVCOLS / 128), T, 0, stream>>>(x_bf, wt_v, bv, v_f, S_LEN, KVCOLS, HDIM);

  // 3) RoPE (Q: no cache; K: writes cache_k), V pass (cache_v + bf16)
  rope_kernel<<<(S_LEN * (HDIM / 2) + T - 1) / T,   T, 0, stream>>>(q_f, q_bf, nullptr, HDIM);
  rope_kernel<<<(S_LEN * (KVCOLS / 2) + T - 1) / T, T, 0, stream>>>(k_f, k_bf, cache_k, KVCOLS);
  {
    size_t n = (size_t)S_LEN * KVCOLS;
    v_pass_kernel<<<(unsigned)((n + T - 1) / T), T, 0, stream>>>(v_f, cache_v, v_bf, n);
  }

  // 4) fused causal attention -> a_bf
  attn_kernel<<<dim3(S_LEN / 128, NHEADS), T, 0, stream>>>(q_bf, k_bf, v_bf, a_bf);

  // 5) output projection -> out (fp32, no bias)
  gemm_bf16_kernel<<<dim3(S_LEN / 128, HDIM / 128), T, 0, stream>>>(a_bf, wt_o, nullptr, out, S_LEN, HDIM, HDIM);
}
